// CustomSchNet_43456479101225
// MI455X (gfx1250) — compile-verified, matrix operand order checked by reading, not run on hardware
//
#include <hip/hip_runtime.h>
#include <math.h>

// ---------------- Model constants (match reference) ----------------
#define HDIM   128        // hidden == filters
#define GDIM   50         // num gaussians
#define GPAD   64         // padded K for WMMA (2 x K32 steps)
#define NATOMS 16384      // 512 molecules * 32 atoms
#define NMOL   512
#define APM    32
#define LINT   3
#define CUTF   6.0f

typedef _Float16 half16 __attribute__((ext_vector_type(16)));
typedef _Float16 half8  __attribute__((ext_vector_type(8)));
typedef float    f32x8  __attribute__((ext_vector_type(8)));

__device__ __forceinline__ float ssp(float x) {
    // shifted softplus, numerically stable
    return fmaxf(x, 0.f) + log1pf(expf(-fabsf(x))) - 0.69314718055994531f;
}

// ---- WMMA fragment loaders (CDNA5 16-bit A 16x32 layout, ISA 7.12.2) ----
// lane<16: elems 0..7 -> K=k0+0..7,  elems 8..15 -> K=k0+16..23 (row M=lane)
// lane>=16: +8 on K, same rows.
__device__ __forceinline__ half16 load_a_f16(const _Float16* M, int stride, int row0, int k0) {
    const int lane = threadIdx.x & 31;
    const _Float16* r = M + (size_t)(row0 + (lane & 15)) * stride + k0 + ((lane >> 4) << 3);
    half8 lo = *(const half8*)r;
    half8 hi = *(const half8*)(r + 16);
    half16 a;
#pragma unroll
    for (int i = 0; i < 8; ++i) { a[i] = lo[i]; a[i + 8] = hi[i]; }
    return a;
}

__device__ __forceinline__ half16 load_a_f32(const float* M, int stride, int row0, int k0) {
    const int lane = threadIdx.x & 31;
    const float* r = M + (size_t)(row0 + (lane & 15)) * stride + k0 + ((lane >> 4) << 3);
    half16 a;
#pragma unroll
    for (int i = 0; i < 8; ++i) { a[i] = (_Float16)r[i]; a[i + 8] = (_Float16)r[i + 16]; }
    return a;
}

// B fragments pre-packed so each lane does a single contiguous 32B load.
__device__ __forceinline__ half16 load_b_pack(const _Float16* P, int ntile, int ktile, int nkt) {
    const int lane = threadIdx.x & 31;
    return *(const half16*)(P + ((((size_t)ntile * nkt + ktile) * 32 + lane) << 4));
}

// ---------------- Weight repack: row-major f32 [K][N] -> WMMA-B f16 ----------------
__global__ void pack_b_kernel(const float* __restrict__ src, _Float16* __restrict__ dst,
                              int Ksrc, int Kpad, int Nn, int nmat, int sstride, int dstride) {
    int idx = blockIdx.x * blockDim.x + threadIdx.x;
    int per = Kpad * Nn;
    if (idx >= per * nmat) return;
    int m = idx / per, r = idx % per;
    int nkt = Kpad >> 5;
    int i = r & 15, lane = (r >> 4) & 31, kt = (r >> 9) % nkt, nt = r / (512 * nkt);
    int n = nt * 16 + (lane & 15);
    int k = kt * 32 + ((lane >> 4) << 3) + (i < 8 ? i : i + 8);
    float v = (k < Ksrc) ? src[(size_t)m * sstride + (size_t)k * Nn + n] : 0.f;
    dst[(size_t)m * dstride + r] = (_Float16)v;
}

// ---------------- h = emb[z] ----------------
__global__ void hinit_kernel(const int* __restrict__ z, const float* __restrict__ emb,
                             float* __restrict__ h) {
    int i = blockIdx.x * blockDim.x + threadIdx.x;
    if (i >= NATOMS * HDIM) return;
    h[i] = emb[(size_t)z[i >> 7] * HDIM + (i & 127)];
}

// ---------------- Edge geometry: distances, gaussian smearing (f16), cosine cutoff ----------------
__global__ __launch_bounds__(256) void edge_geom_kernel(const int* __restrict__ src,
                                                        const int* __restrict__ dst,
                                                        const float* __restrict__ pos,
                                                        int E, int Epad,
                                                        _Float16* __restrict__ ea,
                                                        float* __restrict__ ccut) {
    __shared__ float sd[64];
    const int e0 = blockIdx.x * 64;
    const int t = threadIdx.x;
    if (t < 64) {
        int e = e0 + t;
        float dd = 0.f;
        if (e < E) {
            int a = src[e], b = dst[e];
            float dx = pos[a * 3 + 0] - pos[b * 3 + 0];
            float dy = pos[a * 3 + 1] - pos[b * 3 + 1];
            float dz = pos[a * 3 + 2] - pos[b * 3 + 2];
            dd = sqrtf(dx * dx + dy * dy + dz * dz);
            ccut[e] = 0.5f * (cosf(dd * (3.14159265358979f / CUTF)) + 1.0f);
        } else if (e < Epad) {
            ccut[e] = 0.f;
        }
        sd[t] = dd;
    }
    __syncthreads();
    const int el = t >> 2;
    const int e = e0 + el;
    if (e < Epad) {
        const float dd = sd[el];
        const int g0 = (t & 3) * 16;
        const float step = CUTF / (GDIM - 1);
        const float coeff = -0.5f / (step * step);
        half16 hv;
#pragma unroll
        for (int i = 0; i < 16; ++i) {
            int g = g0 + i;
            float v = 0.f;
            if (g < GDIM && e < E) { float df = dd - (float)g * step; v = expf(coeff * df * df); }
            hv[i] = (_Float16)v;
        }
        *(half16*)(ea + (size_t)e * GPAD + g0) = hv;
    }
}

// ---------------- CSR row pointers: src is globally sorted ----------------
__global__ void rowstart_kernel(const int* __restrict__ src, int E, int* __restrict__ row) {
    int n = blockIdx.x * blockDim.x + threadIdx.x;
    if (n > NATOMS) return;
    int lo = 0, hi = E;
    while (lo < hi) { int mid = (lo + hi) >> 1; if (src[mid] < n) lo = mid + 1; else hi = mid; }
    row[n] = lo;
}

// ---------------- Fused edge-filter MLP: W = ssp(ea@w1+b1)@w2+b2, *Ccut  (all WMMA) ----------------
__global__ __launch_bounds__(256) void edge_mlp_kernel(const _Float16* __restrict__ ea,
                                                       const float* __restrict__ ccut,
                                                       const _Float16* __restrict__ w1p,
                                                       const float* __restrict__ b1,
                                                       const _Float16* __restrict__ w2p,
                                                       const float* __restrict__ b2,
                                                       _Float16* __restrict__ Wout) {
    __shared__ __align__(32) _Float16 s1[16 * HDIM];
    const int w = threadIdx.x >> 5;      // wave id == output n-tile
    const int lane = threadIdx.x & 31;
    const int e0 = blockIdx.x * 16;

    // stage 1: [16,64] @ [64,128] -> this wave's [16,16]
    f32x8 c = {};
#pragma unroll
    for (int kt = 0; kt < 2; ++kt) {
        half16 a = load_a_f16(ea, GPAD, e0, kt * 32);
        half16 b = load_b_pack(w1p, w, kt, 2);
        c = __builtin_amdgcn_wmma_f32_16x16x32_f16(false, a, false, b, (short)0, c, false, false);
    }
    const int n = w * 16 + (lane & 15);
    const float bb1 = b1[n];
#pragma unroll
    for (int r = 0; r < 8; ++r) {
        const int M = r + ((lane >> 4) << 3);
        s1[M * HDIM + n] = (_Float16)ssp(c[r] + bb1);
    }
    __syncthreads();

    // stage 2: [16,128] @ [128,128] -> this wave's [16,16]
    f32x8 c2 = {};
#pragma unroll
    for (int kt = 0; kt < 4; ++kt) {
        half16 a = load_a_f16(s1, HDIM, 0, kt * 32);   // LDS reads (ds_load_b128)
        half16 b = load_b_pack(w2p, w, kt, 4);
        c2 = __builtin_amdgcn_wmma_f32_16x16x32_f16(false, a, false, b, (short)0, c2, false, false);
    }
    const float bb2 = b2[n];
#pragma unroll
    for (int r = 0; r < 8; ++r) {
        const int M = r + ((lane >> 4) << 3);
        float v = (c2[r] + bb2) * ccut[e0 + M];
        Wout[(size_t)(e0 + M) * HDIM + n] = (_Float16)v;
    }
}

// ---------------- Generic node GEMM [N,128]@[128,128] with epilogue modes ----------------
// mode 0: out16 = c            (hx = h@cf1)
// mode 1: out16 = f16(ssp(c+b))(v1 = ssp(agg@cf2+b))
// mode 2: h    += c + b        (residual: h += v1@lin + b)
__global__ __launch_bounds__(256) void node_gemm_kernel(const float* __restrict__ inF,
                                                        const _Float16* __restrict__ inH,
                                                        int in_is_half,
                                                        const _Float16* __restrict__ pack,
                                                        const float* __restrict__ bias,
                                                        _Float16* __restrict__ out16,
                                                        float* __restrict__ h_inout,
                                                        int mode) {
    const int w = threadIdx.x >> 5;
    const int lane = threadIdx.x & 31;
    const int n0 = blockIdx.x * 16;
    f32x8 c = {};
#pragma unroll
    for (int kt = 0; kt < 4; ++kt) {
        half16 a = in_is_half ? load_a_f16(inH, HDIM, n0, kt * 32)
                              : load_a_f32(inF, HDIM, n0, kt * 32);
        half16 b = load_b_pack(pack, w, kt, 4);
        c = __builtin_amdgcn_wmma_f32_16x16x32_f16(false, a, false, b, (short)0, c, false, false);
    }
    const int n = w * 16 + (lane & 15);
    const float bn = bias ? bias[n] : 0.f;
#pragma unroll
    for (int r = 0; r < 8; ++r) {
        const int M = r + ((lane >> 4) << 3);
        const size_t o = (size_t)(n0 + M) * HDIM + n;
        const float v = c[r] + bn;
        if (mode == 0)      out16[o] = (_Float16)v;
        else if (mode == 1) out16[o] = (_Float16)ssp(v);
        else                h_inout[o] += v;
    }
}

// ---------------- Segmented aggregation (no atomics; src-sorted + symmetric graph) ----------------
__global__ __launch_bounds__(128) void aggregate_kernel(const int* __restrict__ dstIdx,
                                                        const int* __restrict__ row,
                                                        const _Float16* __restrict__ hx,
                                                        const _Float16* __restrict__ W,
                                                        float* __restrict__ agg) {
    const int n = blockIdx.x;
    const int f = threadIdx.x;
    const int s = row[n], e = row[n + 1];
    float acc = 0.f;
    for (int i = s; i < e; ++i) {
        int d = dstIdx[i];
        acc += (float)hx[(size_t)d * HDIM + f] * (float)W[(size_t)i * HDIM + f];
    }
    agg[(size_t)n * HDIM + f] = acc;
}

// ---------------- Output head ----------------
__global__ __launch_bounds__(64) void head1_kernel(const float* __restrict__ h,
                                                   const float* __restrict__ w1,
                                                   const float* __restrict__ b1,
                                                   const float* __restrict__ w2,
                                                   const float* __restrict__ b2,
                                                   float* __restrict__ y) {
    __shared__ float red[64];
    const int nid = blockIdx.x;
    const int j = threadIdx.x;
    const float* hr = h + (size_t)nid * HDIM;
    float acc = b1[j];
    for (int k = 0; k < HDIM; ++k) acc = fmaf(hr[k], w1[k * 64 + j], acc);
    red[j] = ssp(acc) * w2[j];
    __syncthreads();
    for (int s = 32; s > 0; s >>= 1) { if (j < s) red[j] += red[j + s]; __syncthreads(); }
    if (j == 0) y[nid] = red[0] + b2[0];
}

__global__ void head2_kernel(const float* __restrict__ y, float* __restrict__ out) {
    int m = blockIdx.x * blockDim.x + threadIdx.x;
    if (m >= NMOL) return;
    float acc = 0.f;
    for (int i = 0; i < APM; ++i) acc += y[m * APM + i];
    out[m] = acc;
}

// ---------------- Orchestration ----------------
extern "C" void kernel_launch(void* const* d_in, const int* in_sizes, int n_in,
                              void* d_out, int out_size, void* d_ws, size_t ws_size,
                              hipStream_t stream) {
    const int*   z    = (const int*)d_in[0];
    const float* pos  = (const float*)d_in[1];
    const int*   ei   = (const int*)d_in[3];
    const float* emb  = (const float*)d_in[4];
    const float* w1   = (const float*)d_in[5];
    const float* b1   = (const float*)d_in[6];
    const float* w2   = (const float*)d_in[7];
    const float* b2   = (const float*)d_in[8];
    const float* cf1  = (const float*)d_in[9];
    const float* cf2  = (const float*)d_in[10];
    const float* cf2b = (const float*)d_in[11];
    const float* lin  = (const float*)d_in[12];
    const float* linb = (const float*)d_in[13];
    const float* o1w  = (const float*)d_in[14];
    const float* o1b  = (const float*)d_in[15];
    const float* o2w  = (const float*)d_in[16];
    const float* o2b  = (const float*)d_in[17];
    float* out = (float*)d_out;
    (void)n_in; (void)out_size; (void)ws_size;

    const int E = in_sizes[3] / 2;
    const int Epad = (E + 15) & ~15;
    const int* src = ei;
    const int* dst = ei + E;

    char* ws = (char*)d_ws;
    size_t off = 0;
    auto take = [&](size_t bytes) -> char* {
        char* p = ws + off;
        off = (off + bytes + 255) & ~(size_t)255;
        return p;
    };
    float*    h    = (float*)take((size_t)NATOMS * HDIM * 4);
    _Float16* hx   = (_Float16*)take((size_t)NATOMS * HDIM * 2);   // also reused for v1
    float*    agg  = (float*)take((size_t)NATOMS * HDIM * 4);
    float*    y    = (float*)take((size_t)NATOMS * 4);
    int*      row  = (int*)take((size_t)(NATOMS + 1) * 4);
    _Float16* w1p  = (_Float16*)take((size_t)LINT * GPAD * HDIM * 2);
    _Float16* w2p  = (_Float16*)take((size_t)LINT * HDIM * HDIM * 2);
    _Float16* cf1p = (_Float16*)take((size_t)LINT * HDIM * HDIM * 2);
    _Float16* cf2p = (_Float16*)take((size_t)LINT * HDIM * HDIM * 2);
    _Float16* linp = (_Float16*)take((size_t)LINT * HDIM * HDIM * 2);
    float*    ccut = (float*)take((size_t)Epad * 4);
    _Float16* ea   = (_Float16*)take((size_t)Epad * GPAD * 2);
    _Float16* Wbuf = (_Float16*)take((size_t)Epad * HDIM * 2);

    // --- one-time prep (cheap, re-run every call for determinism) ---
    {
        int tot = LINT * GPAD * HDIM;
        pack_b_kernel<<<(tot + 255) / 256, 256, 0, stream>>>(w1, w1p, GDIM, GPAD, HDIM, LINT,
                                                             GDIM * HDIM, GPAD * HDIM);
    }
    {
        int tot = LINT * HDIM * HDIM;
        int g = (tot + 255) / 256;
        pack_b_kernel<<<g, 256, 0, stream>>>(w2,  w2p,  HDIM, HDIM, HDIM, LINT, HDIM * HDIM, HDIM * HDIM);
        pack_b_kernel<<<g, 256, 0, stream>>>(cf1, cf1p, HDIM, HDIM, HDIM, LINT, HDIM * HDIM, HDIM * HDIM);
        pack_b_kernel<<<g, 256, 0, stream>>>(cf2, cf2p, HDIM, HDIM, HDIM, LINT, HDIM * HDIM, HDIM * HDIM);
        pack_b_kernel<<<g, 256, 0, stream>>>(lin, linp, HDIM, HDIM, HDIM, LINT, HDIM * HDIM, HDIM * HDIM);
    }
    hinit_kernel<<<(NATOMS * HDIM + 255) / 256, 256, 0, stream>>>(z, emb, h);
    edge_geom_kernel<<<(Epad + 63) / 64, 256, 0, stream>>>(src, dst, pos, E, Epad, ea, ccut);
    rowstart_kernel<<<(NATOMS + 1 + 255) / 256, 256, 0, stream>>>(src, E, row);

    // --- interaction layers ---
    for (int l = 0; l < LINT; ++l) {
        // hx = h @ cf1
        node_gemm_kernel<<<NATOMS / 16, 256, 0, stream>>>(h, (const _Float16*)nullptr, 0,
                                                          cf1p + (size_t)l * HDIM * HDIM, nullptr,
                                                          hx, nullptr, 0);
        // W = ssp(ea@w1+b1)@w2+b2, * Ccut
        edge_mlp_kernel<<<Epad / 16, 256, 0, stream>>>(ea, ccut,
                                                       w1p + (size_t)l * GPAD * HDIM, b1 + l * HDIM,
                                                       w2p + (size_t)l * HDIM * HDIM, b2 + l * HDIM,
                                                       Wbuf);
        // agg[n] = sum over src-run of hx[dst]*W  (symmetric graph identity)
        aggregate_kernel<<<NATOMS, 128, 0, stream>>>(dst, row, hx, Wbuf, agg);
        // v1 = ssp(agg@cf2 + b)  (reuses hx buffer)
        node_gemm_kernel<<<NATOMS / 16, 256, 0, stream>>>(agg, nullptr, 0,
                                                          cf2p + (size_t)l * HDIM * HDIM, cf2b + l * HDIM,
                                                          hx, nullptr, 1);
        // h += v1@lin + b
        node_gemm_kernel<<<NATOMS / 16, 256, 0, stream>>>(nullptr, hx, 1,
                                                          linp + (size_t)l * HDIM * HDIM, linb + l * HDIM,
                                                          nullptr, h, 2);
    }

    // --- output head + per-molecule sum ---
    head1_kernel<<<NATOMS, 64, 0, stream>>>(h, o1w, o1b, o2w, o2b, y);
    head2_kernel<<<(NMOL + 255) / 256, 256, 0, stream>>>(y, out);
}
